// EntropywithDis_54176717472278
// MI455X (gfx1250) — matrix operand order repaired and checked by reading, not compiled, not run on hardware
//
#include <hip/hip_runtime.h>
#include <hip/hip_bf16.h>
#include <math.h>

typedef __attribute__((ext_vector_type(16))) __bf16 v16bf;
typedef __attribute__((ext_vector_type(8)))  float  v8f;
typedef __attribute__((ext_vector_type(4)))  unsigned int v4u;

#define DEG_F      0.017453292519943295f
#define EARTH_R_F  6371.0f
#define NOISE_STD_F (2500.0f / 111320.0f)

#define AST 20   // A LDS row stride in dwords (128 rows)
#define BST 36   // B LDS row stride in dwords (32 k-rows)

// ---------- helpers ----------
__device__ __forceinline__ unsigned short f2bf(float x) {
    unsigned int u = __float_as_uint(x);
    unsigned int r = u + 0x7FFFu + ((u >> 16) & 1u);
    return (unsigned short)(r >> 16);
}

// generic pointer derived from __shared__ carries the LDS byte offset in its low 32 bits
__device__ __forceinline__ unsigned lds_off(const void* p) {
    return (unsigned)(uintptr_t)p;
}

// gfx1250 async global->LDS copy, tracked by ASYNCcnt (no VGPR round-trip)
__device__ __forceinline__ void async_ld_b64(unsigned ldsoff, const void* g) {
    asm volatile("global_load_async_to_lds_b64 %0, %1, off"
                 :: "v"(ldsoff), "v"(g) : "memory");
}

// ---------- 0: queue = gps_queue ----------
__global__ void copy_queue_kernel(const float* __restrict__ src, float* __restrict__ dst, int n) {
    int i = blockIdx.x * blockDim.x + threadIdx.x;
    if (i < n) dst[i] = src[i];
}

// ---------- 1: hard-negative mining ----------
__global__ void mine_kernel(const float* __restrict__ gps,
                            const float* __restrict__ gallery,
                            const int* __restrict__ pool_idx,
                            const int* __restrict__ far_sel,
                            const int* __restrict__ perm,
                            float* __restrict__ queue,
                            int P, int n_near, int n_far, int near_cnt, int per_neg) {
    __shared__ float sd[256];
    __shared__ int   si[256];
    int b = blockIdx.x;
    int t = threadIdx.x;

    float lat1 = gps[b * 2 + 0] * DEG_F;
    float lon1 = gps[b * 2 + 1] * DEG_F;

    float d = 3.0e38f;
    if (t < P) {
        int g = pool_idx[b * P + t];
        float lat2 = gallery[(size_t)g * 2 + 0] * DEG_F;
        float lon2 = gallery[(size_t)g * 2 + 1] * DEG_F;
        float sdl = sinf((lat2 - lat1) * 0.5f);
        float sdo = sinf((lon2 - lon1) * 0.5f);
        float h = sdl * sdl + cosf(lat1) * cosf(lat2) * sdo * sdo;
        h = fminf(fmaxf(h, 0.0f), 1.0f);
        d = 2.0f * EARTH_R_F * asinf(sqrtf(h));
    }
    sd[t] = d;
    si[t] = t;
    __syncthreads();

    for (int k = 2; k <= 256; k <<= 1) {
        for (int j = k >> 1; j > 0; j >>= 1) {
            int ixj = t ^ j;
            if (ixj > t) {
                float d0 = sd[t], d1 = sd[ixj];
                int   i0 = si[t], i1 = si[ixj];
                bool asc = ((t & k) == 0);
                bool gt  = (d0 > d1) || (d0 == d1 && i0 > i1);
                if (gt == asc) {
                    sd[t] = d1; sd[ixj] = d0;
                    si[t] = i1; si[ixj] = i0;
                }
            }
            __syncthreads();
        }
    }

    if (t < per_neg) {
        int pos = (t < n_near) ? t : (near_cnt + far_sel[b * n_far + (t - n_near)]);
        int pslot = si[pos];
        int g = pool_idx[b * P + pslot];
        float la = gallery[(size_t)g * 2 + 0];
        float lo = gallery[(size_t)g * 2 + 1];
        int dq = perm[b * per_neg + t];
        queue[(size_t)dq * 2 + 0] = la;
        queue[(size_t)dq * 2 + 1] = lo;
    }
}

// ---------- 2: threefry2x32-20 noise ----------
__device__ __forceinline__ void threefry2x32(unsigned k0, unsigned k1,
                                             unsigned& x0, unsigned& x1) {
    const unsigned R0[4] = {13u, 15u, 26u, 6u};
    const unsigned R1[4] = {17u, 29u, 16u, 24u};
    unsigned ks[3] = {k0, k1, k0 ^ k1 ^ 0x1BD11BDAu};
    x0 += ks[0]; x1 += ks[1];
#pragma unroll
    for (int i = 0; i < 5; ++i) {
#pragma unroll
        for (int j = 0; j < 4; ++j) {
            unsigned r = (i & 1) ? R1[j] : R0[j];
            x0 += x1;
            x1 = (x1 << r) | (x1 >> (32u - r));
            x1 ^= x0;
        }
        x0 += ks[(i + 1) % 3];
        x1 += ks[(i + 2) % 3] + (unsigned)(i + 1);
    }
}

__global__ void noise_kernel(float* __restrict__ queue, int total) {
    int i = blockIdx.x * blockDim.x + threadIdx.x;
    if (i >= total) return;
    int half = total >> 1;
    unsigned x0, x1;
    if (i < half) { x0 = (unsigned)i;          x1 = (unsigned)(i + half); }
    else          { x0 = (unsigned)(i - half); x1 = (unsigned)i; }
    threefry2x32(0u, 1u, x0, x1);
    unsigned bits = (i < half) ? x0 : x1;
    float f = __uint_as_float((bits >> 9) | 0x3f800000u) - 1.0f;
    float lo = nextafterf(-1.0f, 0.0f);
    float u = fmaxf(f * (1.0f - lo) + lo, lo);
    float n = 1.4142135623730951f * erfinvf(u);
    queue[i] += n * NOISE_STD_F;
}

// ---------- 3: featurize ----------
__global__ void featurize_kernel(const float* __restrict__ gps,
                                 const float* __restrict__ queue,
                                 const float* __restrict__ freqs,
                                 unsigned short* __restrict__ ff,
                                 int B, int F) {
    int row = blockIdx.x;
    int f   = threadIdx.x;
    float lat, lon;
    if (row < B) { lat = gps[row * 2]; lon = gps[row * 2 + 1]; }
    else         { lat = queue[(size_t)(row - B) * 2]; lon = queue[(size_t)(row - B) * 2 + 1]; }
    float ang = lat * freqs[f] + lon * freqs[F + f];
    float s, c;
    __sincosf(ang, &s, &c);
    size_t base = (size_t)row * (2 * F);
    ff[base + f]     = f2bf(s);
    ff[base + F + f] = f2bf(c);
}

// ---------- 4: f32 -> bf16 ----------
__global__ void cvt_bf16_kernel(const float* __restrict__ in, unsigned short* __restrict__ out, size_t n) {
    size_t i = (size_t)blockIdx.x * blockDim.x + threadIdx.x;
    if (i < n) out[i] = f2bf(in[i]);
}

// ---------- 5: bf16 WMMA GEMM, 128x64 block tile, 8 waves, 32x32 per wave ----------
// Double-buffered: async global->LDS batch for step i+1 is issued before the
// WMMAs for step i; s_wait_asynccnt <= (next batch size) exploits in-order
// async completion so only the *previous* batch is required to have landed.
// LDS layouts are fragment-ready:
//   A row m (AST=20 dwords): dwords [0:3]=k0..7, [4:7]=k16..23, [8:11]=k8..15, [12:15]=k24..31
//   B row k (BST=36 dwords): dword v packs n=2v (lo) | n=2v+1 (hi)
// so each A fragment is two aligned b128 LDS reads per lane (B: 16 b16 reads).
template<int BNT, int RELU, int OUTBF>
__global__ void gemm_kernel(const unsigned short* __restrict__ A,
                            const unsigned short* __restrict__ B,
                            float* __restrict__ Cf,
                            unsigned short* __restrict__ Cbf,
                            const float* __restrict__ bias,
                            const float* __restrict__ scale_ptr,
                            int M, int N, int K) {
    __shared__ __align__(16) unsigned int As[2][128 * AST];
    __shared__ __align__(16) unsigned int Bs[2][32 * BST];

    const int t     = threadIdx.x;   // 256 threads = 8 waves
    const int lane  = t & 31;
    const int wave  = t >> 5;
    const int waveM = wave >> 1;     // 0..3 : rows 32*waveM .. +31
    const int waveN = wave & 1;      // 0..1 : cols 32*waveN .. +31
    const int blockM = blockIdx.y * 128;
    const int blockN = blockIdx.x * 64;

    v8f acc[2][2] = {{ {}, {} }, { {}, {} }};

    auto stage = [&](int buf, int k0) {
        // A: 128x32 bf16 = 1024 async-B64 chunks (4 per thread)
#pragma unroll
        for (int i = 0; i < 4; ++i) {
            int c   = t + 256 * i;       // 0..1023
            int row = c >> 3;            // 0..127
            int kq  = c & 7;             // chunk covers k = 4*kq .. 4*kq+3
            int dw  = 2 * (kq & 1) + 8 * ((kq >> 1) & 1) + 4 * (kq >> 2);
            async_ld_b64(lds_off(&As[buf][row * AST + dw]),
                         A + (size_t)(blockM + row) * K + k0 + 4 * kq);
        }
        if (!BNT) {
            // B: 32x64 bf16 = 512 async-B64 chunks (2 per thread)
#pragma unroll
            for (int i = 0; i < 2; ++i) {
                int c = t + 256 * i;     // 0..511
                int k = c >> 4;          // 0..31
                int h = c & 15;          // chunk covers n = 4*h .. 4*h+3
                async_ld_b64(lds_off(&Bs[buf][k * BST + 2 * h]),
                             B + (size_t)(k0 + k) * N + blockN + 4 * h);
            }
        } else {
            // B^T (B is NxK): transpose via b16 LDS stores (DScnt-tracked)
            unsigned short* bs16 = (unsigned short*)&Bs[buf][0];  // (k, n) at k*2*BST + n
#pragma unroll
            for (int i = 0; i < 2; ++i) {
                int c  = t + 256 * i;    // 0..511
                int n  = c >> 3;         // 0..63
                int kq = c & 7;          // covers k = 4*kq .. 4*kq+3
                const unsigned int* g =
                    (const unsigned int*)(B + (size_t)(blockN + n) * K + k0 + 4 * kq);
                unsigned int w0 = g[0], w1 = g[1];
                int kk = 4 * kq;
                bs16[(kk + 0) * (2 * BST) + n] = (unsigned short)(w0 & 0xffffu);
                bs16[(kk + 1) * (2 * BST) + n] = (unsigned short)(w0 >> 16);
                bs16[(kk + 2) * (2 * BST) + n] = (unsigned short)(w1 & 0xffffu);
                bs16[(kk + 3) * (2 * BST) + n] = (unsigned short)(w1 >> 16);
            }
        }
    };

    const int nsteps = K >> 5;
    stage(0, 0);

    for (int i = 0; i < nsteps; ++i) {
        int cur = i & 1;
        if (i + 1 < nsteps) {
            stage(cur ^ 1, 32 * (i + 1));
            // previous batch landed once only the new batch remains in flight
            asm volatile("s_wait_asynccnt %0" :: "i"(BNT ? 4 : 6) : "memory");
        } else {
            asm volatile("s_wait_asynccnt 0x0" ::: "memory");
        }
        __syncthreads();

        union Frag { v4u x[2]; v16bf v; };
        Frag fa[2], fb[2];
        {
            int g  = lane >> 4;                 // K-half select
            int m0 = 32 * waveM + (lane & 15);
#pragma unroll
            for (int mt = 0; mt < 2; ++mt) {
                const unsigned int* p = &As[cur][(m0 + 16 * mt) * AST + 8 * g];
                fa[mt].x[0] = *(const v4u*)(p);
                fa[mt].x[1] = *(const v4u*)(p + 4);
            }
#pragma unroll
            for (int ct = 0; ct < 2; ++ct) {
                const unsigned short* q = (const unsigned short*)&Bs[cur][0] + 32 * waveN + 16 * ct + (lane & 15);
#pragma unroll
                for (int s = 0; s < 16; ++s) {
                    int kk = (s < 8) ? (8 * g + s) : (16 + 8 * g + (s - 8));
                    fb[ct].v[s] = __builtin_bit_cast(__bf16, q[kk * (2 * BST)]);
                }
            }
        }
#pragma unroll
        for (int mt = 0; mt < 2; ++mt)
#pragma unroll
            for (int ct = 0; ct < 2; ++ct)
                acc[mt][ct] = __builtin_amdgcn_wmma_f32_16x16x32_bf16(
                    false, fa[mt].v, false, fb[ct].v, (short)0, acc[mt][ct], false, false);

        // protect the buffer that the next iteration re-stages
        __syncthreads();
    }

    // ---- epilogue: C layout -> lanes 0-15: N=lane, M=r; lanes 16-31: N=lane-16, M=8+r
    float scale = scale_ptr ? scale_ptr[0] : 1.0f;
    int n_local = lane & 15;
    int mhalf   = (lane >> 4) * 8;
#pragma unroll
    for (int ct = 0; ct < 2; ++ct) {
        int col = blockN + 32 * waveN + 16 * ct + n_local;
        float bv = bias ? bias[col] : 0.0f;
#pragma unroll
        for (int mt = 0; mt < 2; ++mt) {
            int rbase = blockM + 32 * waveM + 16 * mt + mhalf;
#pragma unroll
            for (int r = 0; r < 8; ++r) {
                float v = acc[mt][ct][r] * scale + bv;
                if (RELU) v = fmaxf(v, 0.0f);
                size_t o = (size_t)(rbase + r) * N + col;
                if (OUTBF) Cbf[o] = f2bf(v);
                else       Cf[o]  = v;
            }
        }
    }
}

// ---------- 6: row l2-normalize ----------
__global__ void l2norm_kernel(const float* __restrict__ X, unsigned short* __restrict__ Y, int D) {
    __shared__ float red[256];
    int r = blockIdx.x;
    float s = 0.0f;
    for (int c = threadIdx.x; c < D; c += 256) {
        float x = X[(size_t)r * D + c];
        s += x * x;
    }
    red[threadIdx.x] = s;
    __syncthreads();
    for (int o = 128; o > 0; o >>= 1) {
        if (threadIdx.x < o) red[threadIdx.x] += red[threadIdx.x + o];
        __syncthreads();
    }
    float inv = rsqrtf(red[0]);
    for (int c = threadIdx.x; c < D; c += 256)
        Y[(size_t)r * D + c] = f2bf(X[(size_t)r * D + c] * inv);
}

// ---------- 7: per-row (diag - max - log(sumexp)) ----------
__global__ void row_lse_kernel(const float* __restrict__ logits, float* __restrict__ rowv, int N) {
    __shared__ float mred[256];
    __shared__ float sred[256];
    int r = blockIdx.x;
    int t = threadIdx.x;
    float m = -3.0e38f, s = 0.0f;
    for (int c = t; c < N; c += 256) {
        float x = logits[(size_t)r * N + c];
        if (x > m) { s = s * expf(m - x) + 1.0f; m = x; }
        else       { s += expf(x - m); }
    }
    mred[t] = m; sred[t] = s;
    __syncthreads();
    for (int o = 128; o > 0; o >>= 1) {
        if (t < o) {
            float m2 = mred[t + o], s2 = sred[t + o];
            float mm = fmaxf(mred[t], m2);
            sred[t]  = sred[t] * expf(mred[t] - mm) + s2 * expf(m2 - mm);
            mred[t]  = mm;
        }
        __syncthreads();
    }
    if (t == 0)
        rowv[r] = logits[(size_t)r * N + r] - mred[0] - logf(sred[0]);
}

// ---------- 8: loss ----------
__global__ void final_loss_kernel(const float* __restrict__ rowv, float* __restrict__ out, int B) {
    __shared__ float red[256];
    float s = 0.0f;
    for (int i = threadIdx.x; i < B; i += 256) s += rowv[i];
    red[threadIdx.x] = s;
    __syncthreads();
    for (int o = 128; o > 0; o >>= 1) {
        if (threadIdx.x < o) red[threadIdx.x] += red[threadIdx.x + o];
        __syncthreads();
    }
    if (threadIdx.x == 0) out[0] = -red[0] / (float)B;
}

// ---------- launch ----------
extern "C" void kernel_launch(void* const* d_in, const int* in_sizes, int n_in,
                              void* d_out, int out_size, void* d_ws, size_t ws_size,
                              hipStream_t stream) {
    const float* imgs      = (const float*)d_in[0];
    const float* gps       = (const float*)d_in[1];
    const float* gps_queue = (const float*)d_in[2];
    const float* gallery   = (const float*)d_in[3];
    const float* W_img     = (const float*)d_in[4];
    const float* freqs     = (const float*)d_in[5];
    const float* W1        = (const float*)d_in[6];
    const float* b1        = (const float*)d_in[7];
    const float* W2        = (const float*)d_in[8];
    const float* b2        = (const float*)d_in[9];
    const float* logit_sc  = (const float*)d_in[10];
    const int*   pool_idx  = (const int*)d_in[11];
    const int*   far_sel   = (const int*)d_in[12];
    const int*   perm      = (const int*)d_in[13];

    const int B     = in_sizes[1] / 2;            // 512
    const int Q     = in_sizes[2] / 2;            // 16384
    const int D_IMG = in_sizes[0] / B;            // 2048
    const int E     = in_sizes[4] / D_IMG;        // 512
    const int F     = in_sizes[5] / 2;            // 256
    const int H     = in_sizes[6] / (2 * F);      // 1024
    const int P     = in_sizes[11] / B;           // 160
    const int n_far = in_sizes[12] / B;           // 16
    const int per_neg  = Q / B;                   // 32
    const int n_near   = per_neg - n_far;         // 16
    const int near_cnt = P - (int)(P * 0.7);      // 48
    const int M_all    = B + Q;                   // 16896

    char* base = (char*)d_ws;
    size_t off = 0;
    auto alloc = [&](size_t bytes) -> char* {
        char* p = base + off;
        off += (bytes + 255) & ~(size_t)255;
        return p;
    };
    float*          queue    = (float*)alloc((size_t)Q * 2 * 4);
    unsigned short* ffbf     = (unsigned short*)alloc((size_t)M_all * 2 * F * 2);
    unsigned short* imgs_bf  = (unsigned short*)alloc((size_t)B * D_IMG * 2);
    unsigned short* Wimg_bf  = (unsigned short*)alloc((size_t)D_IMG * E * 2);
    unsigned short* W1_bf    = (unsigned short*)alloc((size_t)(2 * F) * H * 2);
    unsigned short* W2_bf    = (unsigned short*)alloc((size_t)H * E * 2);
    float*          emb_raw  = (float*)alloc((size_t)B * E * 4);
    unsigned short* imgemb_bf= (unsigned short*)alloc((size_t)B * E * 2);
    unsigned short* h_bf     = (unsigned short*)alloc((size_t)M_all * H * 2);
    float*          g_f32    = (float*)alloc((size_t)M_all * E * 4);
    unsigned short* gpsemb_bf= (unsigned short*)alloc((size_t)M_all * E * 2);
    float*          logits   = (float*)alloc((size_t)B * M_all * 4);
    float*          rowv     = (float*)alloc((size_t)B * 4);

    copy_queue_kernel<<<(2 * Q + 255) / 256, 256, 0, stream>>>(gps_queue, queue, 2 * Q);
    mine_kernel<<<B, 256, 0, stream>>>(gps, gallery, pool_idx, far_sel, perm, queue,
                                       P, n_near, n_far, near_cnt, per_neg);
    noise_kernel<<<(2 * Q + 255) / 256, 256, 0, stream>>>(queue, 2 * Q);
    featurize_kernel<<<M_all, F, 0, stream>>>(gps, queue, freqs, ffbf, B, F);
    {
        size_t n;
        n = (size_t)B * D_IMG;
        cvt_bf16_kernel<<<(unsigned)((n + 255) / 256), 256, 0, stream>>>(imgs, imgs_bf, n);
        n = (size_t)D_IMG * E;
        cvt_bf16_kernel<<<(unsigned)((n + 255) / 256), 256, 0, stream>>>(W_img, Wimg_bf, n);
        n = (size_t)(2 * F) * H;
        cvt_bf16_kernel<<<(unsigned)((n + 255) / 256), 256, 0, stream>>>(W1, W1_bf, n);
        n = (size_t)H * E;
        cvt_bf16_kernel<<<(unsigned)((n + 255) / 256), 256, 0, stream>>>(W2, W2_bf, n);
    }
    // emb_raw = imgs @ W_img            (512 x 2048 x 512)   NN, f32 out
    gemm_kernel<0, 0, 0><<<dim3(E / 64, B / 128), 256, 0, stream>>>(
        imgs_bf, Wimg_bf, emb_raw, nullptr, nullptr, nullptr, B, E, D_IMG);
    l2norm_kernel<<<B, 256, 0, stream>>>(emb_raw, imgemb_bf, E);
    // h = relu(ff @ W1 + b1)            (16896 x 512 x 1024) NN, bf16 out
    gemm_kernel<0, 1, 1><<<dim3(H / 64, M_all / 128), 256, 0, stream>>>(
        ffbf, W1_bf, nullptr, h_bf, b1, nullptr, M_all, H, 2 * F);
    // g = h @ W2 + b2                   (16896 x 1024 x 512) NN, f32 out
    gemm_kernel<0, 0, 0><<<dim3(E / 64, M_all / 128), 256, 0, stream>>>(
        h_bf, W2_bf, g_f32, nullptr, b2, nullptr, M_all, E, H);
    l2norm_kernel<<<M_all, 256, 0, stream>>>(g_f32, gpsemb_bf, E);
    // logits = scale * img_emb @ gps_emb^T  (512 x 512 x 16896) NT, f32 out
    gemm_kernel<1, 0, 0><<<dim3(M_all / 64, B / 128), 256, 0, stream>>>(
        imgemb_bf, gpsemb_bf, logits, nullptr, nullptr, logit_sc, B, M_all, E);
    row_lse_kernel<<<B, 256, 0, stream>>>(logits, rowv, M_all);
    final_loss_kernel<<<1, 256, 0, stream>>>(rowv, (float*)d_out, B);
}
